// PolarEncoder_61727269978298
// MI455X (gfx1250) — 8-bit matrix operand order repaired and checked by reading, compiled, not run on hardware
//
#include <hip/hip_runtime.h>
#include <hip/hip_bf16.h>

typedef int   v16i __attribute__((ext_vector_type(16)));
typedef float v8f  __attribute__((ext_vector_type(8)));

#define NN 1024
#define KK 512
#define FP8_ONE 0x38u

__global__ void polar_genB(const int* __restrict__ info_pos,
                           unsigned int* __restrict__ wsB) {
  unsigned int U     = blockIdx.x * blockDim.x + threadIdx.x;
  unsigned int w     = U & 3u;
  unsigned int lane  = (U >> 2) & 31u;
  unsigned int q     = (U >> 7) & 3u;
  unsigned int ko    = (U >> 9) & 3u;
  unsigned int ntile = U >> 11;
  unsigned int col   = ntile * 16u + (lane & 15u);
  unsigned int khalf = (lane >> 4) & 1u;
  unsigned int packed = 0u;
#pragma unroll
  for (unsigned int b = 0; b < 4; ++b) {
    unsigned int k = ko * 128u + q * 32u + khalf * 16u + w * 4u + b;
    unsigned int p = (unsigned int)info_pos[k];
    unsigned int v = ((col & ~p) == 0u) ? FP8_ONE : 0u;
    packed |= v << (8u * b);
  }
  wsB[U] = packed;
}

__global__ __launch_bounds__(512)
void polar_encode_wmma(const float* __restrict__ u,
                       const int4* __restrict__ wsB4,
                       float* __restrict__ out) {
  extern __shared__ __align__(16) char smem[];
  unsigned int* ldsA = (unsigned int*)smem;
  int4*         ldsB = (int4*)(smem + 32 * 1024);

  const int tid     = threadIdx.x;
  const int rowbase = blockIdx.x * 64;

  const float4* src4 = (const float4*)(u + (size_t)rowbase * KK);
#pragma unroll
  for (int i = 0; i < 16; ++i) {
    int idx = i * 512 + tid;
    float4 f = src4[idx];
    unsigned int pk;
    pk  =  (f.x != 0.0f) ? FP8_ONE : 0u;
    pk |= ((f.y != 0.0f) ? FP8_ONE : 0u) << 8;
    pk |= ((f.z != 0.0f) ? FP8_ONE : 0u) << 16;
    pk |= ((f.w != 0.0f) ? FP8_ONE : 0u) << 24;
    ldsA[idx] = pk;
  }

  auto issue_plane = [&](int ko, int buf) {
#pragma unroll
    for (int i = 0; i < 16; ++i) {
      int e     = i * 512 + tid;
      int ntile = e >> 7;
      int rem   = e & 127;
      unsigned goff    = (unsigned)((((ntile * 4 + ko) * 128) + rem) * 16);
      unsigned ldsaddr = (unsigned)(size_t)(const void*)(ldsB + buf * 8192 + e);
      asm volatile("global_load_async_to_lds_b128 %0, %1, %2"
                   :: "v"(ldsaddr), "v"(goff), "s"(wsB4) : "memory");
    }
  };

  issue_plane(0, 0);

  const int lane  = tid & 31;
  const int wave  = tid >> 5;
  const int ng    = wave & 7;
  const int mg    = wave >> 3;
  const int m     = lane & 15;
  const int khalf = (lane >> 4) & 1;

  v8f acc[2][8];
#pragma unroll
  for (int mt = 0; mt < 2; ++mt)
#pragma unroll
    for (int nt = 0; nt < 8; ++nt)
#pragma unroll
      for (int j = 0; j < 8; ++j) acc[mt][nt][j] = 0.0f;

#pragma unroll
  for (int ko = 0; ko < 4; ++ko) {
    asm volatile("s_wait_asynccnt 0x0" ::: "memory");
    __syncthreads();
    if (ko < 3) issue_plane(ko + 1, (ko + 1) & 1);

    const int buf = ko & 1;

    v16i a0, a1;
#pragma unroll
    for (int t = 0; t < 8; ++t) {
      int base = ko * 32 + (t >> 1) * 8 + khalf * 4 + (t & 1) * 2;
      uint2 d0 = *(const uint2*)&ldsA[(mg * 32 + m) * 128 + base];
      uint2 d1 = *(const uint2*)&ldsA[(mg * 32 + 16 + m) * 128 + base];
      a0[2 * t] = (int)d0.x;  a0[2 * t + 1] = (int)d0.y;
      a1[2 * t] = (int)d1.x;  a1[2 * t + 1] = (int)d1.y;
    }
#pragma unroll
    for (int nt = 0; nt < 8; ++nt) {
      const int ntile = ng * 8 + nt;
      v16i b;
#pragma unroll
      for (int q = 0; q < 4; ++q) {
        int4 d = ldsB[buf * 8192 + ntile * 128 + q * 32 + lane];
        b[4 * q + 0] = d.x;
        b[4 * q + 1] = d.y;
        b[4 * q + 2] = d.z;
        b[4 * q + 3] = d.w;
      }
      acc[0][nt] = __builtin_amdgcn_wmma_f32_16x16x128_fp8_fp8(
          a0, b, (short)0, acc[0][nt], false, false);
      acc[1][nt] = __builtin_amdgcn_wmma_f32_16x16x128_fp8_fp8(
          a1, b, (short)0, acc[1][nt], false, false);
    }
  }

#pragma unroll
  for (int mt = 0; mt < 2; ++mt) {
#pragma unroll
    for (int nt = 0; nt < 8; ++nt) {
      const int col = (ng * 8 + nt) * 16 + m;
#pragma unroll
      for (int r = 0; r < 8; ++r) {
        const int row = rowbase + mg * 32 + mt * 16 + r + 8 * khalf;
        const int iv  = (int)acc[mt][nt][r];
        out[(size_t)row * NN + col] = (float)(iv & 1);
      }
    }
  }
}

extern "C" void kernel_launch(void* const* d_in, const int* in_sizes, int n_in,
                              void* d_out, int out_size, void* d_ws, size_t ws_size,
                              hipStream_t stream) {
  const float* u        = (const float*)d_in[0];
  const int*   info_pos = (const int*)d_in[1];
  float* out = (float*)d_out;

  const int batch = in_sizes[0] / KK;

  polar_genB<<<512, 256, 0, stream>>>(info_pos, (unsigned int*)d_ws);

  const size_t lds_bytes = 32 * 1024 + 2 * 128 * 1024;
  polar_encode_wmma<<<batch / 64, 512, lds_bytes, stream>>>(
      u, (const int4*)d_ws, out);
}
